// MCMoE_37391985279670
// MI455X (gfx1250) — compile-verified, matrix operand order checked by reading, not run on hardware
//
#include <hip/hip_runtime.h>
#include <hip/hip_bf16.h>

#define DIMD 256
#define NN1  16384
#define NN2  4096
#define ATTD 128

typedef __attribute__((ext_vector_type(16))) _Float16 v16h;
typedef __attribute__((ext_vector_type(8)))  _Float16 v8h;
typedef __attribute__((ext_vector_type(8)))  float    v8f;

// ---- WMMA fragment helpers (CDNA5 16x16x32 f16, wave32) ----------------
// A (16x32): lane&15 = row, halves 0..7 = K{base..base+7}, 8..15 = K{base+16..+23};
//            lanes>=16 shift K base by +8.  B mirrors with lane&15 = col.
// Works for any [16][stride] f16 array (global or LDS).
__device__ __forceinline__ v16h load_frag(const _Float16* base, int stride, int kc, int lane) {
    int r  = lane & 15;
    int ko = kc * 32 + ((lane & 16) ? 8 : 0);
    const _Float16* p = base + (size_t)r * stride + ko;
    v8h lo = *(const v8h*)(p);
    v8h hi = *(const v8h*)(p + 16);
    return __builtin_shufflevector(lo, hi, 0,1,2,3,4,5,6,7,8,9,10,11,12,13,14,15);
}

__device__ __forceinline__ v8f wmma16(v16h a, v16h b, v8f c) {
    return __builtin_amdgcn_wmma_f32_16x16x32_f16(false, a, false, b, (short)0, c, false, false);
}

__device__ __forceinline__ float hmax16(float v) {
    #pragma unroll
    for (int m = 1; m < 16; m <<= 1) v = fmaxf(v, __shfl_xor(v, m, 32));
    return v;
}
__device__ __forceinline__ float hsum16(float v) {
    #pragma unroll
    for (int m = 1; m < 16; m <<= 1) v += __shfl_xor(v, m, 32);
    return v;
}

// Async global->LDS copy of 16B per lane (CDNA5, tracked by ASYNCcnt).
__device__ __forceinline__ void async_stage16(const _Float16* gsrc, unsigned lds_off) {
    asm volatile("global_load_async_to_lds_b128 %0, %1, off"
                 :: "v"(lds_off), "v"(gsrc) : "memory");
}

// ---- small kernels ------------------------------------------------------
__global__ void __launch_bounds__(256) colsum_kernel(const float* __restrict__ x,
                                                     int rows_per_block, int nrows,
                                                     float* __restrict__ out) {
    int tid = threadIdx.x;
    int r0 = blockIdx.x * rows_per_block;
    int r1 = r0 + rows_per_block; if (r1 > nrows) r1 = nrows;
    float s = 0.f;
    for (int r = r0; r < r1; ++r) s += x[(size_t)r * DIMD + tid];
    atomicAdd(&out[tid], s);
}

__global__ void gate_kernel(const float* __restrict__ sim, const float* __restrict__ gatesIn,
                            float* __restrict__ wsf) {
    int lane = threadIdx.x;           // 32 threads
    const float* f1 = wsf;
    const float* f2 = wsf + 256;
    float* gate = wsf + 512;
    float f[8]; float ssq = 0.f;
    #pragma unroll
    for (int i = 0; i < 8; ++i) {
        int d = lane * 8 + i;
        f[i] = 0.5f * (f1[d] / (float)NN1 + f2[d] / (float)NN2);
        ssq += f[i] * f[i];
    }
    #pragma unroll
    for (int m = 1; m < 32; m <<= 1) ssq += __shfl_xor(ssq, m, 32);
    float fs = rsqrtf(ssq + 1e-8f);
    float sc[4];
    #pragma unroll
    for (int e = 0; e < 4; ++e) {
        float dq = 0.f, dd = 0.f;
        #pragma unroll
        for (int i = 0; i < 8; ++i) {
            float sv = sim[e * DIMD + lane * 8 + i];
            dq += f[i] * sv; dd += sv * sv;
        }
        #pragma unroll
        for (int m = 1; m < 32; m <<= 1) { dq += __shfl_xor(dq, m, 32); dd += __shfl_xor(dd, m, 32); }
        sc[e] = dq * fs * rsqrtf(dd + 1e-8f);
    }
    if (lane == 0) {
        float v1 = -1e30f, v2 = -1e30f;
        for (int e = 0; e < 4; ++e) { float s = sc[e]; if (s > v1) { v2 = v1; v1 = s; } else if (s > v2) v2 = s; }
        float w[4]; int cnt = 0;
        for (int e = 0; e < 4; ++e) {
            bool keep = (sc[e] >= v2) && (sc[e] > gatesIn[e]);
            w[e] = keep ? sc[e] : 0.f;
            if (w[e] > 0.f) cnt++;
        }
        float inv = 1.f / (float)(cnt < 1 ? 1 : cnt);
        gate[0] = w[0]; gate[1] = w[1]; gate[2] = w[2]; gate[3] = w[3];
        gate[4] = inv;  gate[5] = w[0] + w[2] + w[3];
    }
}

// W[K][N] f32 -> Wt[N][K] f16 (so B-fragments are contiguous per lane)
__global__ void __launch_bounds__(256) wcvt_kernel(const float* __restrict__ W,
                                                   _Float16* __restrict__ Wt, int Kd, int Nd) {
    int idx = blockIdx.x * 256 + threadIdx.x;
    if (idx >= Kd * Nd) return;
    int k = idx / Nd, n = idx % Nd;
    Wt[(size_t)n * Kd + k] = (_Float16)W[idx];
}

// ---- x2 projections: K16, V16T, SNN2 colsum, DAMISL logits --------------
__global__ void __launch_bounds__(128) proj_x2_kernel(
    const float* __restrict__ x2, const float* __restrict__ g2,
    const float* __restrict__ b2, const float* __restrict__ wa,
    const _Float16* __restrict__ wkT, const _Float16* __restrict__ wvT,
    const _Float16* __restrict__ w2T, const _Float16* __restrict__ vaT,
    const _Float16* __restrict__ uaT,
    _Float16* __restrict__ K16, _Float16* __restrict__ V16T,
    float* __restrict__ snnsum, float* __restrict__ dlog) {
    __shared__ __align__(16) _Float16 Araw[16 * 256];
    __shared__ __align__(16) _Float16 Anrm[16 * 256];
    __shared__ float ssq[16][8];
    int tid = threadIdx.x;
    int m0 = blockIdx.x * 16;
    int row = tid >> 3, seg = tid & 7;
    const float* xr = x2 + (size_t)(m0 + row) * DIMD + seg * 32;
    float vals[32]; float acc = 0.f;
    #pragma unroll
    for (int i = 0; i < 32; ++i) { vals[i] = xr[i]; acc += vals[i] * vals[i]; }
    ssq[row][seg] = acc;
    __syncthreads();
    float rs = 0.f;
    #pragma unroll
    for (int i = 0; i < 8; ++i) rs += ssq[row][i];
    float scale = rsqrtf(rs / 256.f + 1e-6f);
    #pragma unroll
    for (int i = 0; i < 32; ++i) {
        int c = seg * 32 + i;
        Araw[row * 256 + c] = (_Float16)vals[i];
        Anrm[row * 256 + c] = (_Float16)(vals[i] * scale * g2[c]);
    }
    __syncthreads();
    int wave = tid >> 5, lane = tid & 31;
    if (wave == 0) {                           // K16 = x2 @ wk
        for (int t = 0; t < 16; ++t) {
            v8f c = {};
            #pragma unroll
            for (int kc = 0; kc < 8; ++kc)
                c = wmma16(load_frag(Araw, 256, kc, lane), load_frag(wkT + t * 16 * 256, 256, kc, lane), c);
            int col = t * 16 + (lane & 15); int rb = (lane & 16) ? 8 : 0;
            #pragma unroll
            for (int r = 0; r < 8; ++r)
                K16[(size_t)(m0 + rb + r) * DIMD + col] = (_Float16)c[r];
        }
    } else if (wave == 1) {                    // V16T[d][key] = (x2 @ wv)^T
        for (int t = 0; t < 16; ++t) {
            v8f c = {};
            #pragma unroll
            for (int kc = 0; kc < 8; ++kc)
                c = wmma16(load_frag(Araw, 256, kc, lane), load_frag(wvT + t * 16 * 256, 256, kc, lane), c);
            int col = t * 16 + (lane & 15); int rb = (lane & 16) ? 8 : 0;
            #pragma unroll
            for (int r = 0; r < 8; ++r)
                V16T[(size_t)col * NN2 + m0 + rb + r] = (_Float16)c[r];
        }
    } else if (wave == 2) {                    // SNN2: elu(rms(x2)@w2+b2), sum over rows
        for (int t = 0; t < 16; ++t) {
            v8f c = {};
            #pragma unroll
            for (int kc = 0; kc < 8; ++kc)
                c = wmma16(load_frag(Anrm, 256, kc, lane), load_frag(w2T + t * 16 * 256, 256, kc, lane), c);
            int col = t * 16 + (lane & 15);
            float bb = b2[col];
            float part = 0.f;
            #pragma unroll
            for (int r = 0; r < 8; ++r) {
                float v = c[r] + bb;
                part += (v > 0.f) ? v : (expf(v) - 1.f);
            }
            part += __shfl_xor(part, 16, 32);   // rows 0..7 + 8..15
            if (lane < 16) atomicAdd(&snnsum[col], part);
        }
    } else {                                   // DAMISL logits: (tanh(x2@va)*sig(x2@ua))@wa
        v8f racc = {};
        for (int t = 0; t < 8; ++t) {
            v8f cv = {}, cu = {};
            #pragma unroll
            for (int kc = 0; kc < 8; ++kc) {
                v16h a = load_frag(Araw, 256, kc, lane);
                cv = wmma16(a, load_frag(vaT + t * 16 * 256, 256, kc, lane), cv);
                cu = wmma16(a, load_frag(uaT + t * 16 * 256, 256, kc, lane), cu);
            }
            float wav = wa[t * 16 + (lane & 15)];
            #pragma unroll
            for (int r = 0; r < 8; ++r) {
                float h = tanhf(cv[r]) * (1.f / (1.f + expf(-cu[r])));
                racc[r] += h * wav;
            }
        }
        #pragma unroll
        for (int r = 0; r < 8; ++r) {
            float s = hsum16(racc[r]);
            if ((lane & 15) == 0) dlog[m0 + ((lane & 16) ? 8 : 0) + r] = s;
        }
    }
}

// ---- x1 projection: Q16 = f16(x1 @ wq) ----------------------------------
__global__ void __launch_bounds__(128) proj_x1_kernel(
    const float* __restrict__ x1, const _Float16* __restrict__ wqT,
    _Float16* __restrict__ Q16) {
    __shared__ __align__(16) _Float16 A[16 * 256];
    int tid = threadIdx.x;
    int m0 = blockIdx.x * 16;
    for (int i = tid; i < 16 * 256; i += 128) A[i] = (_Float16)x1[(size_t)m0 * DIMD + i];
    __syncthreads();
    int wave = tid >> 5, lane = tid & 31;
    for (int tt = 0; tt < 4; ++tt) {
        int t = wave * 4 + tt;
        v8f c = {};
        #pragma unroll
        for (int kc = 0; kc < 8; ++kc)
            c = wmma16(load_frag(A, 256, kc, lane), load_frag(wqT + t * 16 * 256, 256, kc, lane), c);
        int col = t * 16 + (lane & 15); int rb = (lane & 16) ? 8 : 0;
        #pragma unroll
        for (int r = 0; r < 8; ++r)
            Q16[(size_t)(m0 + rb + r) * DIMD + col] = (_Float16)c[r];
    }
}

// ---- flash attention: AO16 = softmax(Q K^T / 16) V ----------------------
// Double-buffered LDS K chunks filled by CDNA5 async global->LDS loads
// (ASYNCcnt), prefetching chunk ch+1 while WMMAs consume chunk ch.
__global__ void __launch_bounds__(128) attn_kernel(
    const _Float16* __restrict__ Q16, const _Float16* __restrict__ K16,
    const _Float16* __restrict__ V16T, _Float16* __restrict__ AO16) {
    __shared__ __align__(16) _Float16 Ksm[2][32 * 256];  // 2 x 16KB key chunks
    __shared__ __align__(16) _Float16 Psm[4][16 * 32];   // per-wave P transpose buffer
    int tid = threadIdx.x, wave = tid >> 5, lane = tid & 31;
    int m0 = blockIdx.x * 64 + wave * 16;                // each wave: 16 queries
    v16h q[8];
    #pragma unroll
    for (int kc = 0; kc < 8; ++kc) q[kc] = load_frag(Q16 + (size_t)m0 * DIMD, 256, kc, lane);
    v8f O[16] = {};
    float m8[8], l8[8];
    #pragma unroll
    for (int r = 0; r < 8; ++r) { m8[r] = -1e30f; l8[r] = 0.f; }

    const int NCH = NN2 / 32;
    // Each wave async-copies its quarter of a 16KB chunk: 8 instrs x 32 lanes x 16B.
    auto stage_chunk = [&](int kb, int buf) {
        unsigned lbase = (unsigned)(uintptr_t)(&Ksm[buf][0]);
        const _Float16* g = K16 + (size_t)kb * DIMD;
        int c0 = wave * 256 + lane;
        #pragma unroll
        for (int i = 0; i < 8; ++i) {
            int c = c0 + i * 32;                  // 16B unit index within chunk
            async_stage16(g + (size_t)c * 8, lbase + (unsigned)c * 16u);
        }
    };

    stage_chunk(0, 0);                            // prologue: chunk 0 -> buf 0
    for (int ch = 0; ch < NCH; ++ch) {
        int buf = ch & 1;
        int kb = ch * 32;
        if (ch + 1 < NCH) {
            // safe: end-of-iteration barrier of ch-1 means nobody reads buf^1 anymore
            stage_chunk(kb + 32, buf ^ 1);
            asm volatile("s_wait_asynccnt 0x8" ::: "memory");  // chunk ch done (in-order)
        } else {
            asm volatile("s_wait_asynccnt 0x0" ::: "memory");
        }
        __syncthreads();                          // all waves' chunk data resident

        const _Float16* Kc = &Ksm[buf][0];
        v8f s0 = {}, s1 = {};
        #pragma unroll
        for (int kc = 0; kc < 8; ++kc) {
            s0 = wmma16(q[kc], load_frag(Kc, 256, kc, lane), s0);
            s1 = wmma16(q[kc], load_frag(Kc + 16 * 256, 256, kc, lane), s1);
        }
        float alpha[8];
        #pragma unroll
        for (int r = 0; r < 8; ++r) {
            float a0 = s0[r] * 0.0625f, a1 = s1[r] * 0.0625f;   // /sqrt(256)
            float mx = hmax16(fmaxf(a0, a1));
            float nm = fmaxf(m8[r], mx);
            float al = expf(m8[r] - nm);
            float p0 = expf(a0 - nm), p1 = expf(a1 - nm);
            l8[r] = l8[r] * al + hsum16(p0 + p1);
            m8[r] = nm; alpha[r] = al;
            int rowp = r + ((lane & 16) ? 8 : 0);
            int c0 = lane & 15;
            Psm[wave][rowp * 32 + c0]      = (_Float16)p0;
            Psm[wave][rowp * 32 + 16 + c0] = (_Float16)p1;
        }
        #pragma unroll
        for (int n = 0; n < 16; ++n)
            #pragma unroll
            for (int r = 0; r < 8; ++r) O[n][r] *= alpha[r];
        asm volatile("s_wait_dscnt 0x0" ::: "memory");   // P write -> P read (same wave)
        v16h pf = load_frag(&Psm[wave][0], 32, 0, lane);
        #pragma unroll
        for (int n = 0; n < 16; ++n) {
            // V fragment direct from L2-resident V16T[d][key]
            v16h vf = load_frag(V16T + (size_t)(n * 16) * NN2 + kb, NN2, 0, lane);
            O[n] = wmma16(pf, vf, O[n]);
        }
        __syncthreads();                          // everyone done reading buf
    }
    int rb = (lane & 16) ? 8 : 0;
    #pragma unroll
    for (int r = 0; r < 8; ++r) {
        float invl = 1.f / l8[r];
        #pragma unroll
        for (int n = 0; n < 16; ++n)
            AO16[(size_t)(m0 + rb + r) * DIMD + n * 16 + (lane & 15)] = (_Float16)(O[n][r] * invl);
    }
}

// ---- DAMISL finalize + broadcast vector ---------------------------------
__global__ void __launch_bounds__(256) damisl_kernel(
    const float* __restrict__ x2, const float* __restrict__ wf,
    float* __restrict__ wsf) {
    float* gate   = wsf + 512;
    float* snnsum = wsf + 520;
    float* bvec   = wsf + 776;
    float* dlog   = wsf + 1288;
    __shared__ float red[256];
    __shared__ float pool_s[256];
    __shared__ float aprob[NN2];
    int tid = threadIdx.x;
    float mx = -1e30f;
    for (int i = tid; i < NN2; i += 256) mx = fmaxf(mx, dlog[i]);
    red[tid] = mx; __syncthreads();
    for (int s = 128; s > 0; s >>= 1) { if (tid < s) red[tid] = fmaxf(red[tid], red[tid + s]); __syncthreads(); }
    mx = red[0]; __syncthreads();
    float sum = 0.f;
    for (int i = tid; i < NN2; i += 256) sum += expf(dlog[i] - mx);
    red[tid] = sum; __syncthreads();
    for (int s = 128; s > 0; s >>= 1) { if (tid < s) red[tid] += red[tid + s]; __syncthreads(); }
    float inv = 1.f / red[0]; __syncthreads();
    for (int i = tid; i < NN2; i += 256) aprob[i] = expf(dlog[i] - mx) * inv;
    __syncthreads();
    float pd = 0.f;                                  // pooled[d], d = tid
    for (int m = 0; m < NN2; ++m) pd += aprob[m] * x2[(size_t)m * DIMD + tid];
    pool_s[tid] = pd; __syncthreads();
    float fuse = 0.f;                                // (pooled @ wf)[tid]
    for (int d = 0; d < DIMD; ++d) fuse += pool_s[d] * wf[d * DIMD + tid];
    bvec[tid] = gate[1] * (snnsum[tid] / (float)NN2) + gate[2] * fuse;
}

// ---- final: attnout@wo + elu(rms(x1)@w1+b1) + gate combine --------------
__global__ void __launch_bounds__(128) final_kernel(
    const float* __restrict__ x1, const float* __restrict__ g1,
    const float* __restrict__ b1, const _Float16* __restrict__ AO16,
    const _Float16* __restrict__ woT, const _Float16* __restrict__ w1T,
    const float* __restrict__ wsf, float* __restrict__ out) {
    const float* gate = wsf + 512;
    const float* bvec = wsf + 776;
    __shared__ __align__(16) _Float16 A2[16 * 256];
    __shared__ float ssq[16][8];
    int tid = threadIdx.x;
    int m0 = blockIdx.x * 16;
    int row = tid >> 3, seg = tid & 7;
    const float* xr = x1 + (size_t)(m0 + row) * DIMD + seg * 32;
    float vals[32]; float acc = 0.f;
    #pragma unroll
    for (int i = 0; i < 32; ++i) { vals[i] = xr[i]; acc += vals[i] * vals[i]; }
    ssq[row][seg] = acc; __syncthreads();
    float rs = 0.f;
    #pragma unroll
    for (int i = 0; i < 8; ++i) rs += ssq[row][i];
    float scale = rsqrtf(rs / 256.f + 1e-6f);
    #pragma unroll
    for (int i = 0; i < 32; ++i) {
        int c = seg * 32 + i;
        A2[row * 256 + c] = (_Float16)(vals[i] * scale * g1[c]);
    }
    __syncthreads();
    int wave = tid >> 5, lane = tid & 31;
    float w0 = gate[0], w1v = gate[1], invn = gate[4], cx = gate[5];
    for (int tt = 0; tt < 4; ++tt) {
        int t = wave * 4 + tt;
        v8f cwo = {}, csn = {};
        #pragma unroll
        for (int kc = 0; kc < 8; ++kc) {
            cwo = wmma16(load_frag(AO16 + (size_t)m0 * DIMD, 256, kc, lane),
                         load_frag(woT + t * 16 * 256, 256, kc, lane), cwo);
            csn = wmma16(load_frag(A2, 256, kc, lane),
                         load_frag(w1T + t * 16 * 256, 256, kc, lane), csn);
        }
        int col = t * 16 + (lane & 15); int rb = (lane & 16) ? 8 : 0;
        float bb = b1[col], bv = bvec[col];
        #pragma unroll
        for (int r = 0; r < 8; ++r) {
            int m = m0 + rb + r;
            float sn = csn[r] + bb;
            sn = (sn > 0.f) ? sn : (expf(sn) - 1.f);
            float xv = x1[(size_t)m * DIMD + col];
            out[(size_t)m * DIMD + col] = invn * (cx * xv + w0 * cwo[r] + w1v * sn + bv);
        }
    }
}

extern "C" void kernel_launch(void* const* d_in, const int* in_sizes, int n_in,
                              void* d_out, int out_size, void* d_ws, size_t ws_size,
                              hipStream_t stream) {
    const float* x1   = (const float*)d_in[0];
    const float* x2   = (const float*)d_in[1];
    const float* sim  = (const float*)d_in[2];
    const float* gts  = (const float*)d_in[3];
    const float* g1   = (const float*)d_in[4];
    const float* g2   = (const float*)d_in[5];
    const float* snw1 = (const float*)d_in[6];
    const float* snb1 = (const float*)d_in[7];
    const float* snw2 = (const float*)d_in[8];
    const float* snb2 = (const float*)d_in[9];
    const float* wq   = (const float*)d_in[10];
    const float* wk   = (const float*)d_in[11];
    const float* wv   = (const float*)d_in[12];
    const float* wo   = (const float*)d_in[13];
    const float* va   = (const float*)d_in[14];
    const float* ua   = (const float*)d_in[15];
    const float* wa   = (const float*)d_in[16];
    const float* wf   = (const float*)d_in[17];

    float* wsf = (float*)d_ws;
    _Float16* wsh = (_Float16*)((char*)d_ws + 32768);
    _Float16* wqT  = wsh;
    _Float16* wkT  = wqT + 65536;
    _Float16* wvT  = wkT + 65536;
    _Float16* woT  = wvT + 65536;
    _Float16* w1T  = woT + 65536;
    _Float16* w2T  = w1T + 65536;
    _Float16* vaT  = w2T + 65536;          // 128*256
    _Float16* uaT  = vaT + 32768;
    _Float16* Q16  = uaT + 32768;          // 16384*256
    _Float16* K16  = Q16 + (size_t)NN1 * DIMD;
    _Float16* V16T = K16 + (size_t)NN2 * DIMD;
    _Float16* AO16 = V16T + (size_t)NN2 * DIMD;

    hipMemsetAsync(d_ws, 0, 32768, stream);   // zero gate/colsum/snnsum scratch

    colsum_kernel<<<64, 256, 0, stream>>>(x1, 256, NN1, wsf);
    colsum_kernel<<<16, 256, 0, stream>>>(x2, 256, NN2, wsf + 256);
    gate_kernel<<<1, 32, 0, stream>>>(sim, gts, wsf);

    wcvt_kernel<<<256, 256, 0, stream>>>(wq,   wqT, 256, 256);
    wcvt_kernel<<<256, 256, 0, stream>>>(wk,   wkT, 256, 256);
    wcvt_kernel<<<256, 256, 0, stream>>>(wv,   wvT, 256, 256);
    wcvt_kernel<<<256, 256, 0, stream>>>(wo,   woT, 256, 256);
    wcvt_kernel<<<256, 256, 0, stream>>>(snw1, w1T, 256, 256);
    wcvt_kernel<<<256, 256, 0, stream>>>(snw2, w2T, 256, 256);
    wcvt_kernel<<<128, 256, 0, stream>>>(va,   vaT, 256, 128);
    wcvt_kernel<<<128, 256, 0, stream>>>(ua,   uaT, 256, 128);

    proj_x2_kernel<<<NN2 / 16, 128, 0, stream>>>(x2, g2, snb2, wa, wkT, wvT, w2T, vaT, uaT,
                                                 K16, V16T, wsf + 520, wsf + 1288);
    proj_x1_kernel<<<NN1 / 16, 128, 0, stream>>>(x1, wqT, Q16);
    attn_kernel<<<NN1 / 64, 128, 0, stream>>>(Q16, K16, V16T, AO16);
    damisl_kernel<<<1, 256, 0, stream>>>(x2, wf, wsf);
    final_kernel<<<NN1 / 16, 128, 0, stream>>>(x1, g1, snb1, AO16, woT, w1T, wsf, (float*)d_out);
}
